// BahdanauAttention_64390149702441
// MI455X (gfx1250) — matrix operand order repaired and checked by reading, not compiled, not run on hardware
//
#include <hip/hip_runtime.h>
#include <hip/hip_bf16.h>
#include <math.h>
#include <stdint.h>

typedef __bf16 v16bf __attribute__((ext_vector_type(16)));
typedef float  v8f   __attribute__((ext_vector_type(8)));

#define B_ 64
#define S_ 1024
#define D_ 512
#define A_ 512

// ---------------------------------------------------------------------------
// dec_feat[b,a] = dec_hidden[b,:] @ W2[:,a] + b2[a] + b1[a]   (bias fold)
// ---------------------------------------------------------------------------
__global__ __launch_bounds__(256)
void prep_dec(const float* __restrict__ dec, const float* __restrict__ W2,
              const float* __restrict__ b1, const float* __restrict__ b2,
              float* __restrict__ decf) {
    int id = blockIdx.x * 256 + threadIdx.x;   // B_*A_ threads
    int b = id >> 9;
    int a = id & (A_ - 1);
    float s = b1[a] + b2[a];
    const float* dh = dec + b * D_;
    for (int d = 0; d < D_; d += 4) {
        float4 h = *(const float4*)(dh + d);
        s += h.x * W2[(d + 0) * A_ + a];
        s += h.y * W2[(d + 1) * A_ + a];
        s += h.z * W2[(d + 2) * A_ + a];
        s += h.w * W2[(d + 3) * A_ + a];
    }
    decf[id] = s;
}

// ---------------------------------------------------------------------------
// W1T[a,d] = bf16(W1[d,a])  -- transposed bf16 copy so each WMMA B-fragment
// (16 contiguous K values of one column) is one contiguous 32B load.
// ---------------------------------------------------------------------------
__global__ __launch_bounds__(256)
void w1_to_bf16_t(const float* __restrict__ W1, __bf16* __restrict__ W1T) {
    int id = blockIdx.x * 256 + threadIdx.x;   // A_*D_ threads
    int a = id >> 9;
    int d = id & (D_ - 1);
    W1T[(size_t)a * D_ + d] = (__bf16)W1[(size_t)d * A_ + a];
}

// ---------------------------------------------------------------------------
// Fused  e[b,s] = V . tanh(enc[b,s,:] @ W1 + decf[b,:]) + bv
//
// One block = (batch b, 32 rows of S).  2 waves, each owns a 16-row M tile.
// enc tile is DMA'd into LDS with GLOBAL_LOAD_ASYNC_TO_LDS_B128 (ASYNCcnt),
// each wave then builds its 16 bf16 A-fragments ONCE (register-resident for
// the whole N sweep), and K=512 is swept with v_wmma_f32_16x16x32_bf16.
// tanh/V reduction is fused into the epilogue so [B,S,A] never exists.
// ---------------------------------------------------------------------------
__global__ __launch_bounds__(64)
void scores_wmma(const float* __restrict__ enc, const __bf16* __restrict__ W1T,
                 const float* __restrict__ decf, const float* __restrict__ V,
                 const float* __restrict__ bv, float* __restrict__ e_out) {
    __shared__ float sF[32 * 512];   // 64 KB fp32 enc tile; reused for e-reduce

    const int tid = threadIdx.x;
    const int b   = blockIdx.x >> 5;          // 32 S-tiles per batch
    const int s0  = (blockIdx.x & 31) * 32;

    // ---- async DMA: global -> LDS, 16B per lane per op (ASYNCcnt path) ----
    const float* src = enc + (size_t)(b * S_ + s0) * D_;
    unsigned lbase = (unsigned)(uintptr_t)(&sF[0]);   // LDS offset = addr[31:0]
    for (int i = tid * 4; i < 32 * 512; i += 64 * 4) {
        unsigned    laddr = lbase + (unsigned)i * 4u;
        const void* gaddr = (const void*)(src + i);
        asm volatile("global_load_async_to_lds_b128 %0, %1, off"
                     :: "v"(laddr), "v"(gaddr) : "memory");
    }
    asm volatile("s_wait_asynccnt 0x0" ::: "memory");
    __syncthreads();

    const int lane = tid & 31;
    const int wv   = tid >> 5;     // wave 0/1 -> M rows [16*wv, 16*wv+16)
    const int m0   = wv * 16;
    const int hi   = lane >> 4;    // half-wave select
    const int nl   = lane & 15;    // row (A) / col (B) within tile

    // ---- build all 16 A-fragments once (contiguous k order, same as B):
    // lane = row; lanes 0-15 hold K 0..15, lanes 16-31 K 16..31
    v16bf af[16];
    const float* arow = sF + (size_t)(m0 + nl) * 512;
#pragma unroll
    for (int k16 = 0; k16 < 16; ++k16) {
        const int kb = k16 * 32;
        float4 l0 = *(const float4*)(arow + kb + 16 * hi);
        float4 l1 = *(const float4*)(arow + kb + 16 * hi + 4);
        float4 h0 = *(const float4*)(arow + kb + 16 * hi + 8);
        float4 h1 = *(const float4*)(arow + kb + 16 * hi + 12);
        v16bf t;
        t[0]  = (__bf16)l0.x; t[1]  = (__bf16)l0.y;
        t[2]  = (__bf16)l0.z; t[3]  = (__bf16)l0.w;
        t[4]  = (__bf16)l1.x; t[5]  = (__bf16)l1.y;
        t[6]  = (__bf16)l1.z; t[7]  = (__bf16)l1.w;
        t[8]  = (__bf16)h0.x; t[9]  = (__bf16)h0.y;
        t[10] = (__bf16)h0.z; t[11] = (__bf16)h0.w;
        t[12] = (__bf16)h1.x; t[13] = (__bf16)h1.y;
        t[14] = (__bf16)h1.z; t[15] = (__bf16)h1.w;
        af[k16] = t;
    }
    __syncthreads();                  // everyone done reading the fp32 tile
    float* sE = sF;                   // reuse LDS for the e reduction
    if (tid < 32) sE[tid] = 0.0f;
    __syncthreads();

    float ep[8] = {0.f, 0.f, 0.f, 0.f, 0.f, 0.f, 0.f, 0.f};

    for (int n = 0; n < 32; ++n) {
        if (n < 31)  // warm L2/WGP$ for next B panel
            __builtin_prefetch(W1T + (size_t)((n + 1) * 16 + nl) * D_, 0, 0);

        v8f c = {};
#pragma unroll
        for (int k16 = 0; k16 < 16; ++k16) {
            // B fragment (32x16): lanes 0-15 hold K 0-15, lanes 16-31 K 16-31
            // for column nl -> 16 contiguous bf16 from transposed W1.
            v16bf bfr = *(const v16bf*)(W1T + (size_t)(n * 16 + nl) * D_
                                        + k16 * 32 + 16 * hi);
            c = __builtin_amdgcn_wmma_f32_16x16x32_bf16(
                    false, af[k16], false, bfr, (short)0, c, false, false);
        }
        // Epilogue: C element (VGPR r, lane) = row m0+r+8*hi, col n*16+nl.
        const int ac = n * 16 + nl;
        const float dv = decf[b * A_ + ac];
        const float vw = V[ac];
#pragma unroll
        for (int r = 0; r < 8; ++r) ep[r] += tanhf(c[r] + dv) * vw;
    }

    // Reduce partial dot-products across the 16 lanes sharing each row.
#pragma unroll
    for (int r = 0; r < 8; ++r) atomicAdd(&sE[m0 + 8 * hi + r], ep[r]);
    __syncthreads();
    if (tid < 32) e_out[b * S_ + s0 + tid] = sE[tid] + bv[0];
}

// ---------------------------------------------------------------------------
// In-place masked softmax over S (equivalent to softmax*mask renormalized).
// ---------------------------------------------------------------------------
__global__ __launch_bounds__(256)
void softmax_mask(float* __restrict__ e_attn, const float* __restrict__ mask) {
    __shared__ float red[256];
    const int b = blockIdx.x, t = threadIdx.x;
    float* row = e_attn + b * S_;
    const float* mrow = mask + b * S_;

    float v0 = row[t], v1 = row[t + 256], v2 = row[t + 512], v3 = row[t + 768];
    float m = fmaxf(fmaxf(v0, v1), fmaxf(v2, v3));
    red[t] = m; __syncthreads();
    for (int o = 128; o > 0; o >>= 1) {
        if (t < o) red[t] = fmaxf(red[t], red[t + o]);
        __syncthreads();
    }
    m = red[0]; __syncthreads();

    float p0 = __expf(v0 - m) * mrow[t];
    float p1 = __expf(v1 - m) * mrow[t + 256];
    float p2 = __expf(v2 - m) * mrow[t + 512];
    float p3 = __expf(v3 - m) * mrow[t + 768];
    red[t] = p0 + p1 + p2 + p3; __syncthreads();
    for (int o = 128; o > 0; o >>= 1) {
        if (t < o) red[t] += red[t + o];
        __syncthreads();
    }
    const float inv = 1.0f / red[0];
    row[t]       = p0 * inv;
    row[t + 256] = p1 * inv;
    row[t + 512] = p2 * inv;
    row[t + 768] = p3 * inv;
}

// ---------------------------------------------------------------------------
// context[b,d] = sum_s attn[b,s] * enc[b,s,d]   (s split over 4 blocks/batch)
// ---------------------------------------------------------------------------
__global__ __launch_bounds__(256)
void zero_ctx(float* __restrict__ ctx) {
    ctx[blockIdx.x * 256 + threadIdx.x] = 0.0f;
}

__global__ __launch_bounds__(256)
void context_kernel(const float* __restrict__ enc, const float* __restrict__ attn,
                    float* __restrict__ ctx) {
    __shared__ float sw[256];
    const int b = blockIdx.x, sc = blockIdx.y * 256, t = threadIdx.x;
    sw[t] = attn[b * S_ + sc + t];
    __syncthreads();
    float2 acc = make_float2(0.f, 0.f);
    const float* base = enc + (size_t)(b * S_ + sc) * D_ + 2 * t;
    for (int s = 0; s < 256; ++s) {
        float2 ev = *(const float2*)(base + (size_t)s * D_);
        acc.x += sw[s] * ev.x;
        acc.y += sw[s] * ev.y;
    }
    atomicAdd(&ctx[b * D_ + 2 * t + 0], acc.x);
    atomicAdd(&ctx[b * D_ + 2 * t + 1], acc.y);
}

// ---------------------------------------------------------------------------
extern "C" void kernel_launch(void* const* d_in, const int* in_sizes, int n_in,
                              void* d_out, int out_size, void* d_ws, size_t ws_size,
                              hipStream_t stream) {
    const float* dec  = (const float*)d_in[0];  // [B,D]
    const float* enc  = (const float*)d_in[1];  // [B,S,D]
    const float* mask = (const float*)d_in[2];  // [B,S]
    const float* W1   = (const float*)d_in[3];  // [D,A]
    const float* b1   = (const float*)d_in[4];  // [A]
    const float* W2   = (const float*)d_in[5];  // [D,A]
    const float* b2   = (const float*)d_in[6];  // [A]
    const float* V    = (const float*)d_in[7];  // [A,1]
    const float* bv   = (const float*)d_in[8];  // [1]

    float* out_ctx  = (float*)d_out;            // [B,D]
    float* out_attn = out_ctx + B_ * D_;        // [B,S] (holds raw e first)

    float*  decf = (float*)d_ws;                           // B*A f32   (128 KB)
    __bf16* W1T  = (__bf16*)((char*)d_ws + B_ * A_ * 4);   // A*D bf16  (512 KB)

    prep_dec<<<(B_ * A_) / 256, 256, 0, stream>>>(dec, W2, b1, b2, decf);
    w1_to_bf16_t<<<(A_ * D_) / 256, 256, 0, stream>>>(W1, W1T);
    scores_wmma<<<B_ * (S_ / 32), 64, 0, stream>>>(enc, W1T, decf, V, bv, out_attn);
    softmax_mask<<<B_, 256, 0, stream>>>(out_attn, mask);
    zero_ctx<<<(B_ * D_) / 256, 256, 0, stream>>>(out_ctx);
    context_kernel<<<dim3(B_, 4), 256, 0, stream>>>(enc, out_attn, out_ctx);
}
